// GFCNC_83863531422346
// MI455X (gfx1250) — matrix operand order repaired and checked by reading, not compiled, not run on hardware
//
#include <hip/hip_runtime.h>
#include <cstdint>

// ---------------------------------------------------------------------------
// MI455X (gfx1250) SplineConv graph-FCN forward.
// Heavy math: per-basis scatter (SpMM, L2-resident atomics) + dense f16 WMMA
// GEMM (v_wmma_f32_16x16x32_f16). Everything else is bandwidth glue.
// ---------------------------------------------------------------------------

static constexpr int GN[5] = {73728, 36864, 18432, 9216, 4608};
static constexpr float BN_EPS = 1e-5f;

typedef __attribute__((ext_vector_type(16))) _Float16 v16h;
typedef __attribute__((ext_vector_type(8)))  float    v8f;

// ---- monotone float<->uint encoding for atomic max over floats -------------
__device__ __forceinline__ unsigned fenc(float f) {
  unsigned u = __float_as_uint(f);
  return (u & 0x80000000u) ? ~u : (u | 0x80000000u);
}
__device__ __forceinline__ float fdec(unsigned u) {
  u = (u & 0x80000000u) ? (u & 0x7FFFFFFFu) : ~u;
  return __uint_as_float(u);
}
__device__ __forceinline__ int iclamp(int v, int lo, int hi) {
  return v < lo ? lo : (v > hi ? hi : v);
}

// ============================ elementwise kernels ===========================
__global__ void k_zero(float* p, size_t n) {
  size_t t = (size_t)blockIdx.x * blockDim.x + threadIdx.x;
  if (t < n) p[t] = 0.f;
}
__global__ void k_fillu(unsigned* p, unsigned v, size_t n) {
  size_t t = (size_t)blockIdx.x * blockDim.x + threadIdx.x;
  if (t < n) p[t] = v;
}
__global__ void k_count_dst(const int* __restrict__ ei, float* __restrict__ cnt, int E) {
  int e = blockIdx.x * blockDim.x + threadIdx.x;
  if (e < E) atomicAdd(&cnt[ei[E + e]], 1.f);
}
__global__ void k_invert(float* c, int n) {
  int t = blockIdx.x * blockDim.x + threadIdx.x;
  if (t < n) c[t] = 1.f / fmaxf(c[t], 1.f);
}

// ============================ positions / cartesian =========================
__global__ void k_seg_pos(const float* __restrict__ pos, const int* __restrict__ cl,
                          float* __restrict__ psum, float* __restrict__ pcnt, int Nn) {
  int n = blockIdx.x * blockDim.x + threadIdx.x;
  if (n >= Nn) return;
  int d = cl[n];
  atomicAdd(&psum[2 * d + 0], pos[2 * n + 0]);
  atomicAdd(&psum[2 * d + 1], pos[2 * n + 1]);
  atomicAdd(&pcnt[d], 1.f);
}
__global__ void k_pos_div(float* __restrict__ po, const float* __restrict__ psum,
                          const float* __restrict__ pcnt, int Nc) {
  int n = blockIdx.x * blockDim.x + threadIdx.x;
  if (n >= Nc) return;
  float c = fmaxf(pcnt[n], 1.f);
  po[2 * n + 0] = psum[2 * n + 0] / c;
  po[2 * n + 1] = psum[2 * n + 1] / c;
}
__global__ void k_maxd(const float* __restrict__ pos, const int* __restrict__ ei,
                       unsigned* __restrict__ maxd, int E) {
  int e = blockIdx.x * blockDim.x + threadIdx.x;
  if (e >= E) return;
  int a = ei[e], b = ei[E + e];
  float dx = fabsf(pos[2 * b + 0] - pos[2 * a + 0]);
  float dy = fabsf(pos[2 * b + 1] - pos[2 * a + 1]);
  atomicMax(maxd, __float_as_uint(fmaxf(dx, dy)));  // non-negative: bits monotone
}
// open B-spline basis, degree 1, dim 2 -> 4 (weight, kk) pairs per edge
__global__ void k_basis(const float* __restrict__ pos, const int* __restrict__ ei,
                        const unsigned* __restrict__ maxd, float* __restrict__ wts,
                        int* __restrict__ idx, int E, int k) {
  int e = blockIdx.x * blockDim.x + threadIdx.x;
  if (e >= E) return;
  int a = ei[e], b = ei[E + e];
  float inv = 0.5f / __uint_as_float(*maxd);
  float p0 = (pos[2 * b + 0] - pos[2 * a + 0]) * inv + 0.5f;
  float p1 = (pos[2 * b + 1] - pos[2 * a + 1]) * inv + 0.5f;
  float v0 = p0 * (float)(k - 1), v1 = p1 * (float)(k - 1);
  float f0 = floorf(v0), f1 = floorf(v1);
  int b0 = iclamp((int)f0, 0, k - 1), b1 = iclamp((int)f1, 0, k - 1);
  float fr0 = v0 - f0, fr1 = v1 - f1;
#pragma unroll
  for (int s = 0; s < 4; ++s) {
    int s0 = s & 1, s1 = s >> 1;
    float w8 = (s0 ? fr0 : 1.f - fr0) * (s1 ? fr1 : 1.f - fr1);
    int i0 = iclamp(b0 + s0, 0, k - 1), i1 = iclamp(b1 + s1, 0, k - 1);
    wts[e * 4 + s] = w8;
    idx[e * 4 + s] = i0 + k * i1;
  }
}

// ============================ scatter (SpMM) ================================
__global__ void k_scatter_pass(const float* __restrict__ X, const int* __restrict__ ei,
                               const float* __restrict__ wts, const int* __restrict__ idx,
                               float* __restrict__ Y, int E, int Cin, int kk) {
  size_t t = (size_t)blockIdx.x * blockDim.x + threadIdx.x;
  if (t >= (size_t)E * Cin) return;
  int e = (int)(t / Cin), c = (int)(t % Cin);
  int src = ei[e], dst = ei[E + e];
  float xs = X[(size_t)src * Cin + c];
  float acc = 0.f;
#pragma unroll
  for (int q = 0; q < 4; ++q)
    if (idx[e * 4 + q] == kk) acc += wts[e * 4 + q] * xs;
  if (acc != 0.f) atomicAdd(&Y[(size_t)dst * Cin + c], acc);
}
__global__ void k_scatter_sum(const float* __restrict__ X, const int* __restrict__ ei,
                              float* __restrict__ Y, int E, int Cin) {
  size_t t = (size_t)blockIdx.x * blockDim.x + threadIdx.x;
  if (t >= (size_t)E * Cin) return;
  int e = (int)(t / Cin), c = (int)(t % Cin);
  atomicAdd(&Y[(size_t)ei[E + e] * Cin + c], X[(size_t)ei[e] * Cin + c]);
}

// ============================ f32 -> f16 converts ===========================
__global__ void k_cvt_mean(const float* __restrict__ Y, const float* __restrict__ invc,
                           _Float16* __restrict__ Ah, int Nn, int Cin) {
  size_t t = (size_t)blockIdx.x * blockDim.x + threadIdx.x;
  if (t >= (size_t)Nn * Cin) return;
  Ah[t] = (_Float16)(Y[t] * invc[t / Cin]);
}
__global__ void k_cvt(const float* __restrict__ X, _Float16* __restrict__ Ah, size_t n) {
  size_t t = (size_t)blockIdx.x * blockDim.x + threadIdx.x;
  if (t < n) Ah[t] = (_Float16)X[t];
}

// ======== pack weight [K x Nc] f32 into WMMA B-fragment order (f16) =========
// B frag as packed here (k contiguous, matching A): lane L holds column N=L&15; lanes>=16 cover K=16..31;
// element e within lane = local K. One contiguous 16-half blob per lane.
__global__ void k_pack_b(const float* __restrict__ W, _Float16* __restrict__ Bp,
                         int K, int Nc) {
  int t = blockIdx.x * blockDim.x + threadIdx.x;
  int ktiles = K >> 5, ntiles = Nc >> 4;
  int total = ktiles * ntiles * 32;
  if (t >= total) return;
  int lane = t & 31, tile = t >> 5;
  int nt = tile % ntiles, kt = tile / ntiles;
  int kbase = kt * 32 + 16 * (lane >> 4);
  int col = nt * 16 + (lane & 15);
  _Float16* dst = Bp + (size_t)t * 16;
#pragma unroll 4
  for (int e = 0; e < 16; ++e)
    dst[e] = (_Float16)W[(size_t)(kbase + e) * Nc + col];
}

// ============================ WMMA GEMM =====================================
// C[M,Nc](f32,row-major) (=|+=) A[M,K](f16,row-major) @ Bp(packed fragments).
// 4 waves/block, one 16x16 C tile per wave, K-loop in steps of 32.
__global__ void __launch_bounds__(128)
k_wmma_gemm(const _Float16* __restrict__ A, const _Float16* __restrict__ Bp,
            float* __restrict__ C, int M, int Nc, int K, int beta) {
  const int wave = threadIdx.x >> 5;
  const int lane = threadIdx.x & 31;
  const int mt = blockIdx.x * 4 + wave;
  const int nt = blockIdx.y;
  if (mt * 16 >= M) return;
  const int mrow = mt * 16 + (lane & 15);
  const int half = lane >> 4;
  const int ktiles = K >> 5;
  const int ntiles = Nc >> 4;
  const _Float16* arow = A + (size_t)mrow * K;
  v8f acc = {};
  union { v16h v; uint4 u[2]; } af, bf;
  for (int kt = 0; kt < ktiles; ++kt) {
    const int kb = kt << 5;
    // A frag: lane row = mrow; one contiguous 16-half run (K = kb+16h..kb+16h+15)
    af.u[0] = *(const uint4*)(arow + kb + 16 * half);
    af.u[1] = *(const uint4*)(arow + kb + 16 * half + 8);
    const _Float16* bp = Bp + (((size_t)(kt * ntiles + nt)) * 32 + lane) * 16;
    bf.u[0] = *(const uint4*)(bp);
    bf.u[1] = *(const uint4*)(bp + 8);
    if (kt + 1 < ktiles) __builtin_prefetch(arow + kb + 32, 0, 0);  // global_prefetch_b8
    acc = __builtin_amdgcn_wmma_f32_16x16x32_f16(false, af.v, false, bf.v,
                                                 (short)0, acc, false, false);
  }
  const int col = nt * 16 + (lane & 15);
#pragma unroll
  for (int r = 0; r < 8; ++r) {
    size_t ci = (size_t)(mt * 16 + r + 8 * half) * Nc + col;
    float v = acc[r];
    C[ci] = beta ? (C[ci] + v) : v;
  }
}

// ============================ epilogues =====================================
__global__ void k_bias_act(float* __restrict__ C, const float* __restrict__ bias,
                           int Nn, int Cout, int elu) {
  size_t t = (size_t)blockIdx.x * blockDim.x + threadIdx.x;
  if (t >= (size_t)Nn * Cout) return;
  float v = C[t] + bias[t % Cout];
  if (elu && v < 0.f) v = expf(v) - 1.f;
  C[t] = v;
}

// direct per-edge conv (used when Cin==1 or Cout==1; FLOP-trivial)
__global__ void k_direct_edge(const float* __restrict__ X, const int* __restrict__ ei,
                              const float* __restrict__ wts, const int* __restrict__ idx,
                              const float* __restrict__ W, float* __restrict__ Out,
                              int E, int Cin, int Cout) {
  size_t t = (size_t)blockIdx.x * blockDim.x + threadIdx.x;
  if (t >= (size_t)E * Cout) return;
  int e = (int)(t / Cout), co = (int)(t % Cout);
  int src = ei[e], dst = ei[E + e];
  float acc = 0.f;
#pragma unroll
  for (int q = 0; q < 4; ++q) {
    int kk = idx[e * 4 + q];
    float s = 0.f;
    for (int c = 0; c < Cin; ++c)
      s += X[(size_t)src * Cin + c] * W[((size_t)kk * Cin + c) * Cout + co];
    acc += wts[e * 4 + q] * s;
  }
  atomicAdd(&Out[(size_t)dst * Cout + co], acc);
}
__global__ void k_direct_epi(float* __restrict__ Out, const float* __restrict__ X,
                             const float* __restrict__ root, const float* __restrict__ bias,
                             const float* __restrict__ invc, int Nn, int Cin, int Cout, int elu) {
  size_t t = (size_t)blockIdx.x * blockDim.x + threadIdx.x;
  if (t >= (size_t)Nn * Cout) return;
  int n = (int)(t / Cout), co = (int)(t % Cout);
  float v = Out[t] * invc[n];
  for (int c = 0; c < Cin; ++c) v += X[(size_t)n * Cin + c] * root[c * Cout + co];
  v += bias[co];
  if (elu && v < 0.f) v = expf(v) - 1.f;
  Out[t] = v;
}

// ============================ batchnorm =====================================
__global__ void k_bn_stats(const float* __restrict__ X, float* __restrict__ stats,
                           int Nn, int C) {
  int c = blockIdx.x;
  float s = 0.f, s2 = 0.f;
  for (int n = threadIdx.x; n < Nn; n += blockDim.x) {
    float v = X[(size_t)n * C + c];
    s += v; s2 += v * v;
  }
  __shared__ float sh[256], sh2[256];
  sh[threadIdx.x] = s; sh2[threadIdx.x] = s2;
  __syncthreads();
  for (int o = 128; o > 0; o >>= 1) {
    if ((int)threadIdx.x < o) { sh[threadIdx.x] += sh[threadIdx.x + o]; sh2[threadIdx.x] += sh2[threadIdx.x + o]; }
    __syncthreads();
  }
  if (threadIdx.x == 0) {
    float mu = sh[0] / (float)Nn;
    stats[c] = mu;
    stats[C + c] = sh2[0] / (float)Nn - mu * mu;
  }
}
__global__ void k_bn_apply(float* __restrict__ X, const float* __restrict__ stats,
                           const float* __restrict__ g, const float* __restrict__ b,
                           int Nn, int C) {
  size_t t = (size_t)blockIdx.x * blockDim.x + threadIdx.x;
  if (t >= (size_t)Nn * C) return;
  int c = (int)(t % C);
  X[t] = g[c] * (X[t] - stats[c]) * rsqrtf(stats[C + c] + BN_EPS) + b[c];
}

// ============================ pooling / skips ===============================
__global__ void k_pool_scatter(const float* __restrict__ H, const int* __restrict__ cl,
                               unsigned* __restrict__ P, int Nn, int C) {
  size_t t = (size_t)blockIdx.x * blockDim.x + threadIdx.x;
  if (t >= (size_t)Nn * C) return;
  int n = (int)(t / C), c = (int)(t % C);
  atomicMax(&P[(size_t)cl[n] * C + c], fenc(H[t]));
}
__global__ void k_pool_decode(const unsigned* __restrict__ P, float* __restrict__ O, size_t n) {
  size_t t = (size_t)blockIdx.x * blockDim.x + threadIdx.x;
  if (t < n) O[t] = fdec(P[t]);
}
__global__ void k_gather_add(const float* __restrict__ Hc, const int* __restrict__ cl,
                             const float* __restrict__ S, float* __restrict__ Out, int Nn, int C) {
  size_t t = (size_t)blockIdx.x * blockDim.x + threadIdx.x;
  if (t >= (size_t)Nn * C) return;
  int n = (int)(t / C), c = (int)(t % C);
  Out[t] = Hc[(size_t)cl[n] * C + c] + S[t];
}
__global__ void k_gather2(const float* __restrict__ S2, const int* __restrict__ cl2,
                          const int* __restrict__ cl1, float* __restrict__ Out, int Nn, int C) {
  size_t t = (size_t)blockIdx.x * blockDim.x + threadIdx.x;
  if (t >= (size_t)Nn * C) return;
  int n = (int)(t / C), c = (int)(t % C);
  Out[t] = S2[(size_t)cl2[cl1[n]] * C + c];
}

// ============================ host helpers ==================================
static inline unsigned nblk(size_t n, int tb) { return (unsigned)((n + (size_t)tb - 1) / tb); }
static const int TB = 256;

static void run_gemm(hipStream_t s, const _Float16* Ah, const _Float16* Bp, float* C,
                     int M, int Nc, int K, int beta) {
  dim3 gg((unsigned)((M / 16 + 3) / 4), (unsigned)(Nc / 16));
  k_wmma_gemm<<<gg, dim3(128), 0, s>>>(Ah, Bp, C, M, Nc, K, beta);
}

// SplineConv via per-basis SpMM + dense WMMA GEMM. wts/idx==nullptr => k2==1.
static void spline_conv_gemm(hipStream_t s, const float* X, const int* ei, int E,
                             int Nn, const float* wts, const int* idx, int k2,
                             const float* W, const float* root, const float* bias,
                             const float* invc, int Cin, int Cout, int elu,
                             float* Out, float* Y, _Float16* Ah, _Float16* Bp) {
  size_t nY = (size_t)Nn * Cin;
  size_t nE = (size_t)E * Cin;
  int pbT = (Cin / 32) * (Cout / 16) * 32;
  for (int kk = 0; kk < k2; ++kk) {
    k_zero<<<nblk(nY, TB), TB, 0, s>>>(Y, nY);
    if (k2 > 1)
      k_scatter_pass<<<nblk(nE, TB), TB, 0, s>>>(X, ei, wts, idx, Y, E, Cin, kk);
    else
      k_scatter_sum<<<nblk(nE, TB), TB, 0, s>>>(X, ei, Y, E, Cin);
    k_cvt_mean<<<nblk(nY, TB), TB, 0, s>>>(Y, invc, Ah, Nn, Cin);
    k_pack_b<<<nblk((size_t)pbT, TB), TB, 0, s>>>(W + (size_t)kk * Cin * Cout, Bp, Cin, Cout);
    run_gemm(s, Ah, Bp, Out, Nn, Cout, Cin, kk > 0);
  }
  // root term: Out += X @ root
  k_cvt<<<nblk(nY, TB), TB, 0, s>>>(X, Ah, nY);
  k_pack_b<<<nblk((size_t)pbT, TB), TB, 0, s>>>(root, Bp, Cin, Cout);
  run_gemm(s, Ah, Bp, Out, Nn, Cout, Cin, 1);
  k_bias_act<<<nblk((size_t)Nn * Cout, TB), TB, 0, s>>>(Out, bias, Nn, Cout, elu);
}

static void spline_conv_direct(hipStream_t s, const float* X, const int* ei, int E,
                               int Nn, const float* wts, const int* idx, const float* W,
                               const float* root, const float* bias, const float* invc,
                               int Cin, int Cout, int elu, float* Out) {
  k_zero<<<nblk((size_t)Nn * Cout, TB), TB, 0, s>>>(Out, (size_t)Nn * Cout);
  k_direct_edge<<<nblk((size_t)E * Cout, TB), TB, 0, s>>>(X, ei, wts, idx, W, Out, E, Cin, Cout);
  k_direct_epi<<<nblk((size_t)Nn * Cout, TB), TB, 0, s>>>(Out, X, root, bias, invc, Nn, Cin, Cout, elu);
}

static void bn_run(hipStream_t s, float* H, int Nn, int C, const float* g, const float* b,
                   float* stats) {
  k_bn_stats<<<dim3((unsigned)C), dim3(256), 0, s>>>(H, stats, Nn, C);
  k_bn_apply<<<nblk((size_t)Nn * C, TB), TB, 0, s>>>(H, stats, g, b, Nn, C);
}

static void pool_max(hipStream_t s, const float* H, const int* cl, float* Out,
                     unsigned* enc, int Nin, int Nout, int C) {
  size_t n = (size_t)Nout * C;
  k_fillu<<<nblk(n, TB), TB, 0, s>>>(enc, 0x00800000u /* fenc(-FLT_MAX) */, n);
  k_pool_scatter<<<nblk((size_t)Nin * C, TB), TB, 0, s>>>(H, cl, enc, Nin, C);
  k_pool_decode<<<nblk(n, TB), TB, 0, s>>>(enc, Out, n);
}

static void seg_mean_pos(hipStream_t s, const float* pin, const int* cl, float* pout,
                         float* psum, float* pcnt, int Nin, int Nout) {
  k_zero<<<nblk((size_t)Nout * 2, TB), TB, 0, s>>>(psum, (size_t)Nout * 2);
  k_zero<<<nblk((size_t)Nout, TB), TB, 0, s>>>(pcnt, (size_t)Nout);
  k_seg_pos<<<nblk((size_t)Nin, TB), TB, 0, s>>>(pin, cl, psum, pcnt, Nin);
  k_pos_div<<<nblk((size_t)Nout, TB), TB, 0, s>>>(pout, psum, pcnt, Nout);
}

// ============================ entry point ===================================
extern "C" void kernel_launch(void* const* d_in, const int* in_sizes, int n_in,
                              void* d_out, int out_size, void* d_ws, size_t ws_size,
                              hipStream_t stream) {
  (void)n_in; (void)out_size; (void)ws_size;
  const float* X0   = (const float*)d_in[0];
  const float* POS0 = (const float*)d_in[1];
  const int* EI[5]; const int* CL[4]; int E[5];
  for (int l = 0; l < 5; ++l) { EI[l] = (const int*)d_in[2 + l]; E[l] = in_sizes[2 + l] / 2; }
  for (int l = 0; l < 4; ++l) CL[l] = (const int*)d_in[7 + l];
  // params flattened: conv{w,root,b} x12, then bn{g,b} x4
  const float* PRM[36];
  for (int i = 0; i < 36; ++i) PRM[i] = (const float*)d_in[12 + i];
#define CW(i) PRM[3 * (i) + 0]
#define CR(i) PRM[3 * (i) + 1]
#define CB(i) PRM[3 * (i) + 2]
  const float* BNG[4]; const float* BNB[4];
  for (int i = 0; i < 4; ++i) { BNG[i] = (const float*)d_in[48 + 2 * i]; BNB[i] = (const float*)d_in[49 + 2 * i]; }

  // ---- workspace bump allocator (256B aligned blocks) ----
  char* base = (char*)d_ws; size_t off = 0;
  auto alloc = [&](size_t bytes) -> void* {
    void* p = base + off; off = (off + bytes + 255) & ~(size_t)255; return p;
  };
  float* cnt[5];
  for (int l = 0; l < 5; ++l) cnt[l] = (float*)alloc((size_t)GN[l] * 4);
  float* pos1 = (float*)alloc((size_t)GN[1] * 2 * 4);
  float* pos2 = (float*)alloc((size_t)GN[2] * 2 * 4);
  float* pos3 = (float*)alloc((size_t)GN[3] * 2 * 4);
  float* psum = (float*)alloc((size_t)GN[1] * 2 * 4);
  float* pcnt = (float*)alloc((size_t)GN[1] * 4);
  unsigned* maxd = (unsigned*)alloc(4 * 4);
  float* wts[4]; int* idx[4];
  for (int l = 0; l < 4; ++l) {
    wts[l] = (float*)alloc((size_t)E[l] * 4 * 4);
    idx[l] = (int*)alloc((size_t)E[l] * 4 * 4);
  }
  float* h0   = (float*)alloc((size_t)GN[0] * 32 * 4);
  float* h0b  = (float*)alloc((size_t)GN[0] * 32 * 4);
  float* p1b  = (float*)alloc((size_t)GN[1] * 32 * 4);
  float* h2a  = (float*)alloc((size_t)GN[1] * 64 * 4);
  float* h2b  = (float*)alloc((size_t)GN[1] * 64 * 4);
  float* p2b  = (float*)alloc((size_t)GN[2] * 64 * 4);   // pool2_x (skip)
  float* h3a  = (float*)alloc((size_t)GN[2] * 128 * 4);
  float* h3b  = (float*)alloc((size_t)GN[2] * 128 * 4);
  float* p3b  = (float*)alloc((size_t)GN[3] * 128 * 4);  // pool3_x (skip)
  float* h4a  = (float*)alloc((size_t)GN[3] * 256 * 4);
  float* h4b  = (float*)alloc((size_t)GN[3] * 256 * 4);
  float* p4b  = (float*)alloc((size_t)GN[4] * 256 * 4);
  float* hfr  = (float*)alloc((size_t)GN[4] * 32 * 4);
  float* sp3f = (float*)alloc((size_t)GN[3] * 32 * 4);
  float* s3f  = (float*)alloc((size_t)GN[3] * 32 * 4);
  float* sp2f = (float*)alloc((size_t)GN[2] * 32 * 4);
  float* s2f  = (float*)alloc((size_t)GN[2] * 32 * 4);
  float* s0f  = (float*)alloc((size_t)GN[0] * 32 * 4);
  unsigned* penc = (unsigned*)alloc((size_t)GN[1] * 32 * 4);  // == N2*64 == N3*128 == N4*256
  float* Y = (float*)alloc((size_t)GN[0] * 32 * 4);           // == max Nn*Cin at every level
  _Float16* Ah = (_Float16*)alloc((size_t)GN[0] * 32 * 2);
  _Float16* Bp = (_Float16*)alloc((size_t)256 * 256 * 2);
  float* stats = (float*)alloc(512 * 4);

  // ---- per-level prep: inverse degree, pooled positions, spline basis ----
  for (int l = 0; l < 5; ++l) {
    k_zero<<<nblk((size_t)GN[l], TB), TB, 0, stream>>>(cnt[l], (size_t)GN[l]);
    k_count_dst<<<nblk((size_t)E[l], TB), TB, 0, stream>>>(EI[l], cnt[l], E[l]);
    k_invert<<<nblk((size_t)GN[l], TB), TB, 0, stream>>>(cnt[l], GN[l]);
  }
  seg_mean_pos(stream, POS0, CL[0], pos1, psum, pcnt, GN[0], GN[1]);
  seg_mean_pos(stream, pos1, CL[1], pos2, psum, pcnt, GN[1], GN[2]);
  seg_mean_pos(stream, pos2, CL[2], pos3, psum, pcnt, GN[2], GN[3]);
  const float* posl[4] = {POS0, pos1, pos2, pos3};
  const int kdeg[4] = {5, 3, 3, 3};
  for (int l = 0; l < 4; ++l) {
    k_fillu<<<1, 1, 0, stream>>>(maxd + l, 0u, (size_t)1);
    k_maxd<<<nblk((size_t)E[l], TB), TB, 0, stream>>>(posl[l], EI[l], maxd + l, E[l]);
    k_basis<<<nblk((size_t)E[l], TB), TB, 0, stream>>>(posl[l], EI[l], maxd + l,
                                                       wts[l], idx[l], E[l], kdeg[l]);
  }

  // ---- encoder ----
  spline_conv_direct(stream, X0, EI[0], E[0], GN[0], wts[0], idx[0],
                     CW(0), CR(0), CB(0), cnt[0], 1, 32, 1, h0);                 // conv1a
  spline_conv_gemm(stream, h0, EI[0], E[0], GN[0], wts[0], idx[0], 25,
                   CW(1), CR(1), CB(1), cnt[0], 32, 32, 1, h0b, Y, Ah, Bp);      // conv1b
  bn_run(stream, h0b, GN[0], 32, BNG[0], BNB[0], stats);
  pool_max(stream, h0b, CL[0], p1b, penc, GN[0], GN[1], 32);
  spline_conv_gemm(stream, p1b, EI[1], E[1], GN[1], wts[1], idx[1], 9,
                   CW(2), CR(2), CB(2), cnt[1], 32, 64, 1, h2a, Y, Ah, Bp);      // conv2a
  spline_conv_gemm(stream, h2a, EI[1], E[1], GN[1], wts[1], idx[1], 9,
                   CW(3), CR(3), CB(3), cnt[1], 64, 64, 1, h2b, Y, Ah, Bp);      // conv2b
  bn_run(stream, h2b, GN[1], 64, BNG[1], BNB[1], stats);
  pool_max(stream, h2b, CL[1], p2b, penc, GN[1], GN[2], 64);
  spline_conv_gemm(stream, p2b, EI[2], E[2], GN[2], wts[2], idx[2], 9,
                   CW(4), CR(4), CB(4), cnt[2], 64, 128, 1, h3a, Y, Ah, Bp);     // conv3a
  spline_conv_gemm(stream, h3a, EI[2], E[2], GN[2], nullptr, nullptr, 1,
                   CW(5), CR(5), CB(5), cnt[2], 128, 128, 1, h3b, Y, Ah, Bp);    // conv3b
  bn_run(stream, h3b, GN[2], 128, BNG[2], BNB[2], stats);
  pool_max(stream, h3b, CL[2], p3b, penc, GN[2], GN[3], 128);
  spline_conv_gemm(stream, p3b, EI[3], E[3], GN[3], nullptr, nullptr, 1,
                   CW(6), CR(6), CB(6), cnt[3], 128, 256, 1, h4a, Y, Ah, Bp);    // conv4a
  spline_conv_gemm(stream, h4a, EI[3], E[3], GN[3], nullptr, nullptr, 1,
                   CW(7), CR(7), CB(7), cnt[3], 256, 256, 1, h4b, Y, Ah, Bp);    // conv4b
  bn_run(stream, h4b, GN[3], 256, BNG[3], BNB[3], stats);
  pool_max(stream, h4b, CL[3], p4b, penc, GN[3], GN[4], 256);
  spline_conv_gemm(stream, p4b, EI[4], E[4], GN[4], nullptr, nullptr, 1,
                   CW(8), CR(8), CB(8), cnt[4], 256, 32, 1, hfr, Y, Ah, Bp);     // score_fr

  // ---- FCN-8s decoder: upsample + skips ----
  spline_conv_gemm(stream, p3b, EI[3], E[3], GN[3], wts[3], idx[3], 9,
                   CW(9), CR(9), CB(9), cnt[3], 128, 32, 1, sp3f, Y, Ah, Bp);    // score_pool3
  k_gather_add<<<nblk((size_t)GN[3] * 32, TB), TB, 0, stream>>>(hfr, CL[3], sp3f, s3f, GN[3], 32);
  spline_conv_gemm(stream, p2b, EI[2], E[2], GN[2], wts[2], idx[2], 9,
                   CW(10), CR(10), CB(10), cnt[2], 64, 32, 1, sp2f, Y, Ah, Bp);  // score_pool2
  k_gather_add<<<nblk((size_t)GN[2] * 32, TB), TB, 0, stream>>>(s3f, CL[2], sp2f, s2f, GN[2], 32);
  k_gather2<<<nblk((size_t)GN[0] * 32, TB), TB, 0, stream>>>(s2f, CL[1], CL[0], s0f, GN[0], 32);
  spline_conv_direct(stream, s0f, EI[0], E[0], GN[0], wts[0], idx[0],
                     CW(11), CR(11), CB(11), cnt[0], 32, 1, 0, (float*)d_out);   // convout
#undef CW
#undef CR
#undef CB
}
